// Wireless_Autoencoder_19361712571400
// MI455X (gfx1250) — matrix operand order repaired and checked by reading, not compiled, not run on hardware
//
#include <hip/hip_runtime.h>

// ---------------- constants ----------------
#define B_TOTAL 262144
#define NOISE_SCALE 0.44668359215096315f   // 1/sqrt(2*R*10^(EbNo/10)), R=0.5, EbNo=7dB
#define EPS_BN 1e-5f

typedef _Float16 v16h __attribute__((ext_vector_type(16)));
typedef float    v8f  __attribute__((ext_vector_type(8)));

// ---------------- activations (branchless) ----------------
__device__ __forceinline__ float elu_f(float x) {
  // x>0: max=x, min terms = 0 ; x<=0: max=0, min(exp(x)-1,0)=exp(x)-1
  return fmaxf(x, 0.f) + fminf(__expf(fminf(x, 0.f)) - 1.f, 0.f);
}
template <int ACT>
__device__ __forceinline__ float act_apply(float x) {
  if (ACT == 1) return fmaxf(x, 0.f);   // relu
  if (ACT == 2) return elu_f(x);        // elu
  return x;
}

// ---------------- WMMA fragment builders (branchless) ----------------
// 16-bit A-matrix 16x32 layout (wave32): lane = 16*h + r, r = row M.
//   element e<8  -> K = K0 + 8h + e ; element e>=8 -> K = K0 + 8h + e + 8
__device__ __forceinline__ v16h a_frag(const float* row, int h, int K0, int Kdim) {
  v16h a;
#pragma unroll
  for (int e = 0; e < 16; ++e) {
    int k  = K0 + 8 * h + e + ((e >= 8) ? 8 : 0);
    int ks = (k < Kdim) ? k : 0;             // safe address (cndmask, no branch)
    float v = row[ks];                       // unconditional LDS load
    a[e] = (k < Kdim) ? (_Float16)v : (_Float16)0.0f;
  }
  return a;
}
// B-matrix 32x16 layout: lane = 16*h + n (n = column N); element e<8 -> K = K0 + 8h + e ; e>=8 -> K = K0 + 8h + e + 8.
// B[k][n] = W[n][k]  (computing X @ W^T with W row-major, shape 16 x Kdim)
__device__ __forceinline__ v16h b_frag(const float* W, int n, int h, int K0, int Kdim) {
  v16h b;
#pragma unroll
  for (int e = 0; e < 16; ++e) {
    int k  = K0 + 8 * h + e + ((e >= 8) ? 8 : 0);
    int ks = (k < Kdim) ? k : 0;
    float v = W[n * Kdim + ks];
    b[e] = (k < Kdim) ? (_Float16)v : (_Float16)0.0f;
  }
  return b;
}

// One wave computes Y = act(X @ W^T + bias) for its 32 samples (two 16x16 tiles).
// X rows live in buf (per-wave LDS region, `stride` floats per row, Kdim cols used).
// Result (16 cols) written back to buf rows, or to global `outg` if TOGLOBAL.
template <int KD, int ACT, bool TOGLOBAL>
__device__ __forceinline__ void wave_mm(float* buf, int stride,
                                        const float* W, const float* bias,
                                        int lane, float* outg, int sbase) {
  const int h = (lane >> 4) & 1, n = lane & 15;
  constexpr int NK = (KD + 31) / 32;
  v16h af[2][NK];
  __syncthreads();                       // make prior LDS writes visible
#pragma unroll
  for (int t = 0; t < 2; ++t)
#pragma unroll
    for (int kc = 0; kc < NK; ++kc)
      af[t][kc] = a_frag(buf + (16 * t + n) * stride, h, kc * 32, KD);
  v16h bf[NK];
#pragma unroll
  for (int kc = 0; kc < NK; ++kc) bf[kc] = b_frag(W, n, h, kc * 32, KD);
  const float bn = bias[n];
  __syncthreads();                       // all reads done before overwriting
#pragma unroll
  for (int t = 0; t < 2; ++t) {
    v8f c = {0.f, 0.f, 0.f, 0.f, 0.f, 0.f, 0.f, 0.f};
#pragma unroll
    for (int kc = 0; kc < NK; ++kc)
      c = __builtin_amdgcn_wmma_f32_16x16x32_f16(false, af[t][kc], false, bf[kc],
                                                 (short)0, c, false, false);
#pragma unroll
    for (int i = 0; i < 8; ++i) {        // C/D layout: c[i] = Y[16t + i + 8h][n]
      float v = act_apply<ACT>(c[i] + bn);
      if (TOGLOBAL) outg[(sbase + 16 * t + i + 8 * h) * 16 + n] = v;
      else          buf[(16 * t + i + 8 * h) * stride + n] = v;
    }
  }
  __syncthreads();
}

// Encoder for 32 samples: z = elu(x@ew1^T+eb1)@ew2^T+eb2, left in buf rows (cols 0..15).
__device__ __forceinline__ void encoder32(float* buf, int stride, const float* x, int sbase,
                                          const float* ew1, const float* eb1,
                                          const float* ew2, const float* eb2, int lane) {
#pragma unroll
  for (int i = lane; i < 512; i += 32) {
    int r = i >> 4, f = i & 15;
    buf[r * stride + f] = x[(sbase + r) * 16 + f];
  }
  wave_mm<16, 2, false>(buf, stride, ew1, eb1, lane, nullptr, 0);  // elu
  wave_mm<16, 0, false>(buf, stride, ew2, eb2, lane, nullptr, 0);  // linear
}

// ---------------- kernel 0: zero the BN accumulators ----------------
__global__ void k_zero(float* p) {
  if (threadIdx.x < 32) p[threadIdx.x] = 0.f;
}

// ---------------- kernel 1: encoder + BN partial sums ----------------
__global__ __launch_bounds__(128) void k_enc_stats(const float* __restrict__ x,
                                                   const float* __restrict__ ew1,
                                                   const float* __restrict__ eb1,
                                                   const float* __restrict__ ew2,
                                                   const float* __restrict__ eb2,
                                                   float* gsum, float* gsq) {
  __shared__ float sw[544 + 32];          // ew1(256) ew2(256) eb1(16) eb2(16)
  __shared__ float sbuf[4][32 * 17];
  __shared__ float ssum[16], ssq[16];
  const int tid = threadIdx.x, wave = tid >> 5, lane = tid & 31;
  for (int i = tid; i < 256; i += 128) { sw[i] = ew1[i]; sw[256 + i] = ew2[i]; }
  if (tid < 16) { sw[512 + tid] = eb1[tid]; sw[528 + tid] = eb2[tid]; ssum[tid] = 0.f; ssq[tid] = 0.f; }
  __syncthreads();
  const int sbase = blockIdx.x * 128 + wave * 32;
  float* buf = sbuf[wave];
  encoder32(buf, 17, x, sbase, sw, sw + 512, sw + 256, sw + 528, lane);
  if (lane < 16) {
    float s = 0.f, q = 0.f;
    for (int r = 0; r < 32; ++r) { float v = buf[r * 17 + lane]; s += v; q += v * v; }
    atomicAdd(&ssum[lane], s);
    atomicAdd(&ssq[lane], q);
  }
  __syncthreads();
  if (tid < 16) { atomicAdd(&gsum[tid], ssum[tid]); atomicAdd(&gsq[tid], ssq[tid]); }
}

// ---------------- kernel 2: finalize mu / rstd ----------------
__global__ void k_stats_final(const float* gsum, const float* gsq, float* mu, float* rstd) {
  int t = threadIdx.x;
  if (t < 16) {
    float m = gsum[t] / (float)B_TOTAL;
    float v = gsq[t] / (float)B_TOTAL - m * m;
    mu[t] = m;
    rstd[t] = rsqrtf(v + EPS_BN);
  }
}

// ---------------- kernel 3: full fused pipeline ----------------
struct Params {
  const float* x;
  const float* noise;
  const float* fading;
  const float* w[32];   // d_in[3..34] in order
  const float* mu;
  const float* rstd;
  float* out;
};

// LDS weight-pool offsets (floats), in d_in[3..34] order
#define OFF_EW1 0
#define OFF_EB1 256
#define OFF_EW2 272
#define OFF_EB2 528
#define OFF_BNG 544
#define OFF_BNB 560
#define OFF_PC1W 576
#define OFF_PC1B 640
#define OFF_PC2W 656
#define OFF_PC2B 1168
#define OFF_PC3W 1184
#define OFF_PC3B 1696
#define OFF_PC4W 1712
#define OFF_PC4B 1968
#define OFF_PC5W 1984
#define OFF_PC5B 2752
#define OFF_PLW 2768
#define OFF_PLB 3024
#define OFF_DC1W 3040
#define OFF_DC1B 3048
#define OFF_DC2W 3056
#define OFF_DC2B 3184
#define OFF_DC3W 3192
#define OFF_DC3B 3256
#define OFF_DC4W 3264
#define OFF_DC4B 3328
#define OFF_DL1W 3336
#define OFF_DL1B 4360
#define OFF_DL2W 4376
#define OFF_DL2B 4632
#define OFF_DL3W 4648
#define OFF_DL3B 4904
#define W_TOTAL 4920

__global__ __launch_bounds__(128) void k_pipeline(Params P) {
  __shared__ float s_w[W_TOTAL];
  __shared__ float s_buf[4][32 * 68];
  const int tid = threadIdx.x, wave = tid >> 5, lane = tid & 31;
  const int STR = 68;

  // stage all weights/biases into LDS once per block
  static const int WOFF[32] = {OFF_EW1, OFF_EB1, OFF_EW2, OFF_EB2, OFF_BNG, OFF_BNB,
                               OFF_PC1W, OFF_PC1B, OFF_PC2W, OFF_PC2B, OFF_PC3W, OFF_PC3B,
                               OFF_PC4W, OFF_PC4B, OFF_PC5W, OFF_PC5B, OFF_PLW, OFF_PLB,
                               OFF_DC1W, OFF_DC1B, OFF_DC2W, OFF_DC2B, OFF_DC3W, OFF_DC3B,
                               OFF_DC4W, OFF_DC4B, OFF_DL1W, OFF_DL1B, OFF_DL2W, OFF_DL2B,
                               OFF_DL3W, OFF_DL3B};
  static const int WSZ[32] = {256, 16, 256, 16, 16, 16, 64, 16, 512, 16, 512, 16,
                              256, 16, 768, 16, 256, 16, 8, 8, 128, 8, 64, 8,
                              64, 8, 1024, 16, 256, 16, 256, 16};
  for (int a = 0; a < 32; ++a)
    for (int i = tid; i < WSZ[a]; i += 128) s_w[WOFF[a] + i] = P.w[a][i];
  __syncthreads();

  float* buf = s_buf[wave];
  const int sbase = blockIdx.x * 128 + wave * 32;
  const int s = sbase + lane;  // this lane's sample

  // ---- encoder (recompute; WMMA) ----
  encoder32(buf, STR, P.x, sbase, s_w + OFF_EW1, s_w + OFF_EB1, s_w + OFF_EW2, s_w + OFF_EB2, lane);

  // ---- batchnorm (per-lane) ----
  float zr[16];
#pragma unroll
  for (int f = 0; f < 16; ++f)
    zr[f] = (buf[lane * STR + f] - P.mu[f]) * P.rstd[f] * s_w[OFF_BNG + f] + s_w[OFF_BNB + f];

  // ---- fading channel + AWGN ----
  float hr[3], hi[3];
#pragma unroll
  for (int j = 0; j < 3; ++j) { hr[j] = P.fading[s * 6 + j * 2]; hi[j] = P.fading[s * 6 + j * 2 + 1]; }
  float cre[16];
#pragma unroll
  for (int i = 0; i < 8; ++i) {
    float orr = 0.f, oii = 0.f;
#pragma unroll
    for (int j = 0; j < 3; ++j) {
      int idx = i + j - 2;
      if (idx >= 0 && idx < 8) {
        float cr = zr[2 * idx], ci = zr[2 * idx + 1];
        float fr = hr[2 - j], fi = hi[2 - j];
        orr += cr * fr - ci * fi;
        oii += cr * fi + ci * fr;
      }
    }
    cre[2 * i]     = orr + P.noise[s * 16 + 2 * i] * NOISE_SCALE;
    cre[2 * i + 1] = oii + P.noise[s * 16 + 2 * i + 1] * NOISE_SCALE;
  }

  // ---- pilot conv stack (per-lane, streamed to bound registers) ----
  float p4v[48];  // [ci][t3] = p4v[ci*3+t3]
#pragma unroll
  for (int t3 = 0; t3 < 3; ++t3) {
    float p1c[48];  // p1 columns 4t3, 4t3+1, 4t3+2 : p1c[ci*3+tt]
#pragma unroll
    for (int tt = 0; tt < 3; ++tt) {
      int tcol = 4 * t3 + tt;
      for (int c = 0; c < 16; ++c) {
        float acc = s_w[OFF_PC1B + c];
#pragma unroll
        for (int k = 0; k < 4; ++k) acc += s_w[OFF_PC1W + c * 4 + k] * cre[tcol + k];
        p1c[c * 3 + tt] = fmaxf(acc, 0.f);
      }
    }
    float p2c[32];  // p2 columns 4t3, 4t3+1 : p2c[ci*2+a]
#pragma unroll
    for (int a = 0; a < 2; ++a) {
      for (int c = 0; c < 16; ++c) {
        float acc = s_w[OFF_PC2B + c];
        for (int ci = 0; ci < 16; ++ci)
          acc += s_w[OFF_PC2W + (c * 16 + ci) * 2 + 0] * p1c[ci * 3 + a] +
                 s_w[OFF_PC2W + (c * 16 + ci) * 2 + 1] * p1c[ci * 3 + a + 1];
        p2c[c * 2 + a] = fmaxf(acc, 0.f);
      }
    }
    float p3v[16];
    for (int c = 0; c < 16; ++c) {
      float acc = s_w[OFF_PC3B + c];
      for (int ci = 0; ci < 16; ++ci)
        acc += s_w[OFF_PC3W + (c * 16 + ci) * 2 + 0] * p2c[ci * 2 + 0] +
               s_w[OFF_PC3W + (c * 16 + ci) * 2 + 1] * p2c[ci * 2 + 1];
      p3v[c] = fmaxf(acc, 0.f);
    }
    for (int c = 0; c < 16; ++c) {
      float acc = s_w[OFF_PC4B + c];
      for (int ci = 0; ci < 16; ++ci) acc += s_w[OFF_PC4W + c * 16 + ci] * p3v[ci];
      p4v[c * 3 + t3] = fmaxf(acc, 0.f);
    }
  }
  // pc5 (k=3) -> p5[16], stage for pl-linear WMMA
  for (int c = 0; c < 16; ++c) {
    float acc = s_w[OFF_PC5B + c];
    for (int ci = 0; ci < 16; ++ci)
#pragma unroll
      for (int k = 0; k < 3; ++k) acc += s_w[OFF_PC5W + (c * 16 + ci) * 3 + k] * p4v[ci * 3 + k];
    buf[lane * STR + c] = fmaxf(acc, 0.f);
  }

  // ---- pilot linear hhat = p5 @ plw^T + plb (WMMA) ----
  wave_mm<16, 0, false>(buf, STR, s_w + OFF_PLW, s_w + OFF_PLB, lane, nullptr, 0);

  // ---- equalization t = c / hhat (complex) ----
  float d0[16];
#pragma unroll
  for (int k = 0; k < 8; ++k) {
    float hhr = buf[lane * STR + 2 * k], hhi = buf[lane * STR + 2 * k + 1];
    float den = hhr * hhr + hhi * hhi;
    float ccr = cre[2 * k], cci = cre[2 * k + 1];
    d0[2 * k]     = (ccr * hhr + cci * hhi) / den;
    d0[2 * k + 1] = (cci * hhr - ccr * hhi) / den;
  }

  // ---- decoder conv stack, streamed; d4 flat (c*8+t) -> buf row cols 0..63 ----
#pragma unroll
  for (int t = 0; t < 8; ++t) {
    float x0 = d0[2 * t], x1 = d0[2 * t + 1];
    float d1a[8], d1b[8], d2v[8], d3v[8];
#pragma unroll
    for (int c = 0; c < 8; ++c) {
      float w = s_w[OFF_DC1W + c], b = s_w[OFF_DC1B + c];
      d1a[c] = fmaxf(w * x0 + b, 0.f);
      d1b[c] = fmaxf(w * x1 + b, 0.f);
    }
#pragma unroll
    for (int c = 0; c < 8; ++c) {
      float acc = s_w[OFF_DC2B + c];
#pragma unroll
      for (int ci = 0; ci < 8; ++ci)
        acc += s_w[OFF_DC2W + (c * 8 + ci) * 2 + 0] * d1a[ci] +
               s_w[OFF_DC2W + (c * 8 + ci) * 2 + 1] * d1b[ci];
      d2v[c] = fmaxf(acc, 0.f);
    }
#pragma unroll
    for (int c = 0; c < 8; ++c) {
      float acc = s_w[OFF_DC3B + c];
#pragma unroll
      for (int ci = 0; ci < 8; ++ci) acc += s_w[OFF_DC3W + c * 8 + ci] * d2v[ci];
      d3v[c] = fmaxf(acc, 0.f);
    }
#pragma unroll
    for (int c = 0; c < 8; ++c) {
      float acc = s_w[OFF_DC4B + c];
#pragma unroll
      for (int ci = 0; ci < 8; ++ci) acc += s_w[OFF_DC4W + c * 8 + ci] * d3v[ci];
      buf[lane * STR + c * 8 + t] = fmaxf(acc, 0.f);
    }
  }

  // ---- dense head: dl1 (64->16, relu), dl2 (elu), dl3 (linear, to global) ----
  wave_mm<64, 1, false>(buf, STR, s_w + OFF_DL1W, s_w + OFF_DL1B, lane, nullptr, 0);
  wave_mm<16, 2, false>(buf, STR, s_w + OFF_DL2W, s_w + OFF_DL2B, lane, nullptr, 0);
  wave_mm<16, 0, true >(buf, STR, s_w + OFF_DL3W, s_w + OFF_DL3B, lane, P.out, sbase);
}

// ---------------- host launcher ----------------
extern "C" void kernel_launch(void* const* d_in, const int* in_sizes, int n_in,
                              void* d_out, int out_size, void* d_ws, size_t ws_size,
                              hipStream_t stream) {
  (void)in_sizes; (void)n_in; (void)out_size; (void)ws_size;
  const float* x      = (const float*)d_in[0];
  const float* ew1    = (const float*)d_in[3];
  const float* eb1    = (const float*)d_in[4];
  const float* ew2    = (const float*)d_in[5];
  const float* eb2    = (const float*)d_in[6];

  float* ws   = (float*)d_ws;
  float* gsum = ws;        // 16
  float* gsq  = ws + 16;   // 16
  float* mu   = ws + 32;   // 16
  float* rstd = ws + 48;   // 16

  const int blocks = B_TOTAL / 128;  // 2048, 128 samples per block (4 waves x 32)

  k_zero<<<1, 64, 0, stream>>>(ws);
  k_enc_stats<<<blocks, 128, 0, stream>>>(x, ew1, eb1, ew2, eb2, gsum, gsq);
  k_stats_final<<<1, 32, 0, stream>>>(gsum, gsq, mu, rstd);

  Params P;
  P.x      = x;
  P.noise  = (const float*)d_in[1];
  P.fading = (const float*)d_in[2];
  for (int i = 0; i < 32; ++i) P.w[i] = (const float*)d_in[3 + i];
  P.mu   = mu;
  P.rstd = rstd;
  P.out  = (float*)d_out;
  k_pipeline<<<blocks, 128, 0, stream>>>(P);
}
